// MultiHeadAttentionLayer_24266565222596
// MI455X (gfx1250) — matrix operand order repaired and checked by reading, not compiled, not run on hardware
//
#include <hip/hip_runtime.h>
#include <hip/hip_bf16.h>

// MI455X / gfx1250: wave32, WMMA 16x16x32 bf16 -> f32 accumulate.

typedef __attribute__((ext_vector_type(16))) __bf16 v16bf;
typedef __attribute__((ext_vector_type(8)))  float  v8f;

#define HIDN 1024
#define NH   16
#define HD   64
#define SEQ  1024
#define NB   8

static __device__ __forceinline__ unsigned short f2bfu(float f) {
  unsigned int u = __builtin_bit_cast(unsigned int, f);
  return (unsigned short)((u + 0x7fffu + ((u >> 16) & 1u)) >> 16);  // RNE
}
static __device__ __forceinline__ __bf16 f2bf(float f) {
  unsigned short h = f2bfu(f);
  return __builtin_bit_cast(__bf16, h);
}

// ---------------------------------------------------------------------------
// Elementwise f32 -> bf16 (vectorized: float4 in, 4x bf16 packed u64 out).
// ---------------------------------------------------------------------------
__global__ __launch_bounds__(256)
void cvt_bf16(const float4* __restrict__ src, unsigned long long* __restrict__ dst,
              int n4)
{
  const int i = blockIdx.x * 256 + threadIdx.x;
  if (i < n4) {
    const float4 f = src[i];
    unsigned long long r = (unsigned long long)f2bfu(f.x)
                         | ((unsigned long long)f2bfu(f.y) << 16)
                         | ((unsigned long long)f2bfu(f.z) << 32)
                         | ((unsigned long long)f2bfu(f.w) << 48);
    dst[i] = r;
  }
}

// ---------------------------------------------------------------------------
// GEMM: out[M=8192,N=1024] = A[M,K=1024](bf16) @ W[N,K](bf16)^T + bias, *scale
// OMODE: 0 = bf16 head-major [B,H,S,D], 1 = bf16 head-transposed [B,H,D,S],
//        2 = f32 row-major [M,N]
// Block: 256 threads = 8 waves in a 2x4 grid; each wave owns a 32x32 tile
// (4 accumulators; 2 A-frags x 2 B-frags -> 4 WMMAs per load group).
// A-fragment: interleaved 16-bit A layout (K = 8*half + {0..7,16..23}).
// B-fragment: same interleaved layout as A (K = 8*half + {0..7,16..23}).
// ---------------------------------------------------------------------------
template<int OMODE>
static __device__ __forceinline__
void storeTile(const v8f& acc, int mB, int nB, int half, int l16,
               const float* __restrict__ bias, float scale, void* __restrict__ outp)
{
  const int ncol = nB + l16;
  const float bv = bias[ncol];
#pragma unroll
  for (int r = 0; r < 8; ++r) {
    const int m = mB + r + half * 8;
    const float v = (acc[r] + bv) * scale;
    if (OMODE == 2) {
      ((float*)outp)[(size_t)m * HIDN + ncol] = v;
    } else {
      const int bb = m >> 10, s = m & 1023;
      const int h  = ncol >> 6, d = ncol & 63;
      if (OMODE == 0)
        ((__bf16*)outp)[(((size_t)(bb * NH + h) * SEQ + s) << 6) + d] = f2bf(v);
      else
        ((__bf16*)outp)[(((size_t)(bb * NH + h) * HD + d) << 10) + s] = f2bf(v);
    }
  }
}

template<int OMODE>
__global__ __launch_bounds__(256)
void gemm32(const __bf16* __restrict__ A, const __bf16* __restrict__ Wb,
            const float* __restrict__ bias, float scale, void* __restrict__ outp)
{
  const int lane = threadIdx.x & 31;
  const int wave = threadIdx.x >> 5;
  const int half = lane >> 4;
  const int l16  = lane & 15;
  const int wr = wave >> 2;                 // 0..1
  const int wc = wave & 3;                  // 0..3
  const int mBase = blockIdx.x * 64 + wr * 32;
  const int nBase = blockIdx.y * 128 + wc * 32;
  const int kA = half * 8;
  const int kB = half * 8;

  const __bf16* a0p = A  + (size_t)(mBase + l16) * HIDN + kA;
  const __bf16* a1p = a0p + (size_t)16 * HIDN;
  const __bf16* b0p = Wb + (size_t)(nBase + l16) * HIDN + kB;
  const __bf16* b1p = b0p + (size_t)16 * HIDN;

  v8f acc00 = {0.f,0.f,0.f,0.f,0.f,0.f,0.f,0.f};
  v8f acc01 = acc00, acc10 = acc00, acc11 = acc00;

  for (int kk = 0; kk < HIDN; kk += 32) {
    __builtin_prefetch(a0p + kk + 128, 0, 1);   // global_prefetch_b8
    __builtin_prefetch(b0p + kk + 128, 0, 1);
    v16bf a0, a1, b0, b1;
#pragma unroll
    for (int j = 0; j < 8; ++j) {
      a0[j] = a0p[kk + j];  a0[j + 8] = a0p[kk + 16 + j];
      a1[j] = a1p[kk + j];  a1[j + 8] = a1p[kk + 16 + j];
    }
#pragma unroll
    for (int e = 0; e < 16; ++e) { b0[e] = b0p[kk + e + (e & 8)]; b1[e] = b1p[kk + e + (e & 8)]; }
    acc00 = __builtin_amdgcn_wmma_f32_16x16x32_bf16(false, a0, false, b0, (short)0, acc00, false, false);
    acc01 = __builtin_amdgcn_wmma_f32_16x16x32_bf16(false, a0, false, b1, (short)0, acc01, false, false);
    acc10 = __builtin_amdgcn_wmma_f32_16x16x32_bf16(false, a1, false, b0, (short)0, acc10, false, false);
    acc11 = __builtin_amdgcn_wmma_f32_16x16x32_bf16(false, a1, false, b1, (short)0, acc11, false, false);
  }

  storeTile<OMODE>(acc00, mBase,      nBase,      half, l16, bias, scale, outp);
  storeTile<OMODE>(acc01, mBase,      nBase + 16, half, l16, bias, scale, outp);
  storeTile<OMODE>(acc10, mBase + 16, nBase,      half, l16, bias, scale, outp);
  storeTile<OMODE>(acc11, mBase + 16, nBase + 16, half, l16, bias, scale, outp);
}

// ---------------------------------------------------------------------------
// Fused attention: per block = (16 query rows) x (one head) x (one batch).
// Phase 1: energy = (Qs)(Ks)^T via WMMA, masked, staged to LDS [16][1024].
// Phase 2: block-wide softmax in LDS; f32 probs written back coalesced and a
//          bf16 copy kept in LDS for phase 3.
// Phase 3: ctx = attn @ V via WMMA -- ALL 8 waves: K-dim split across wave
//          pairs (kp halves), partials combined through LDS.
// ---------------------------------------------------------------------------
__global__ __launch_bounds__(256)
void attn_kernel(const __bf16* __restrict__ Qbf, const __bf16* __restrict__ Kbf,
                 const __bf16* __restrict__ Vt, const unsigned char* __restrict__ mask,
                 float* __restrict__ attnOut, __bf16* __restrict__ ctx)
{
  __shared__ float  eT[16][SEQ];    // 64 KB energy/prob stripe (f32)
  __shared__ __bf16 eB[16][SEQ];    // 32 KB prob stripe (bf16) for phase 3
  __shared__ float  red[16][16];    // row reductions
  __shared__ float  part[16][HD];   //  4 KB ctx partials (kp-half 1)

  const int tid  = threadIdx.x;
  const int lane = tid & 31;
  const int wave = tid >> 5;
  const int half = lane >> 4;
  const int l16  = lane & 15;
  const int qt = blockIdx.x;        // query tile 0..63
  const int h  = blockIdx.y;        // head
  const int bb = blockIdx.z;        // batch
  const int qBase = qt * 16;

  const size_t headOff = ((size_t)(bb * NH + h)) * SEQ * HD;
  const __bf16* Qh = Qbf + headOff;                 // [SEQ][HD]
  const __bf16* Kh = Kbf + headOff;                 // [SEQ][HD]
  const __bf16* Vh = Vt  + headOff;                 // [HD][SEQ] (transposed)
  const unsigned char* mrow = mask + (size_t)bb * SEQ;

  const int kA = half * 8;
  const int kB = half * 8;
  const float NEGINF = __builtin_bit_cast(float, 0xff800000u);

  // ---- Phase 1: energy tiles (each wave owns key tiles wave, wave+8, ...) ----
  const __bf16* qrow = Qh + (size_t)(qBase + l16) * HD;
  for (int kt = wave; kt < SEQ / 16; kt += 8) {
    const int kcol = kt * 16 + l16;
    const __bf16* krow = Kh + (size_t)kcol * HD;
    v8f acc = {0.f,0.f,0.f,0.f,0.f,0.f,0.f,0.f};
#pragma unroll
    for (int kk = 0; kk < HD; kk += 32) {
      v16bf a, b;
#pragma unroll
      for (int j = 0; j < 8; ++j) { a[j] = qrow[kk + kA + j]; a[j + 8] = qrow[kk + kA + 16 + j]; }
#pragma unroll
      for (int e = 0; e < 16; ++e) b[e] = krow[kk + kB + e + (e & 8)];
      acc = __builtin_amdgcn_wmma_f32_16x16x32_bf16(false, a, false, b,
                                                    (short)0, acc, false, false);
    }
    const bool msk = (mrow[kcol] != 0);
#pragma unroll
    for (int r = 0; r < 8; ++r)
      eT[r + half * 8][kcol] = msk ? NEGINF : acc[r];
  }
  __syncthreads();

  // ---- Phase 2: softmax over each of the 16 rows (16 threads per row) ----
  const int row = tid >> 4;
  const int seg = tid & 15;
  float lmax = NEGINF;
  for (int c = seg * 64; c < seg * 64 + 64; ++c) lmax = fmaxf(lmax, eT[row][c]);
  red[row][seg] = lmax;
  __syncthreads();
  float rmax = red[row][0];
#pragma unroll
  for (int i = 1; i < 16; ++i) rmax = fmaxf(rmax, red[row][i]);
  __syncthreads();
  float lsum = 0.f;
  for (int c = seg * 64; c < seg * 64 + 64; ++c) {
    float ex = __expf(eT[row][c] - rmax);
    eT[row][c] = ex;
    lsum += ex;
  }
  red[row][seg] = lsum;
  __syncthreads();
  float rsum = 0.f;
#pragma unroll
  for (int i = 0; i < 16; ++i) rsum += red[row][i];
  const float inv = 1.f / rsum;
  for (int c = seg * 64; c < seg * 64 + 64; ++c) {
    const float p = eT[row][c] * inv;
    eT[row][c] = p;
    eB[row][c] = f2bf(p);
  }
  __syncthreads();

  // ---- Writeback attention probabilities (coalesced) ----
  float* aout = attnOut + (((size_t)(bb * NH + h)) * SEQ + qBase) * SEQ;
  for (int idx = tid; idx < 16 * SEQ; idx += 256)
    aout[idx] = eT[idx >> 10][idx & 1023];

  // ---- Phase 3: ctx = attn @ V, all 8 waves ----
  // wave = hk*4 + dt: dt selects the 16-wide d tile, hk selects the kp half.
  const int dt = wave & 3;
  const int hk = wave >> 2;
  const int d  = dt * 16 + l16;
  const __bf16* vrow = Vh + (size_t)d * SEQ;   // K-contiguous thanks to V^T
  v8f acc = {0.f,0.f,0.f,0.f,0.f,0.f,0.f,0.f};
  const int kp0 = hk * (SEQ / 2);
  for (int kp = kp0; kp < kp0 + SEQ / 2; kp += 32) {
    v16bf a, b;
#pragma unroll
    for (int j = 0; j < 8; ++j) {
      a[j]     = eB[l16][kp + kA + j];
      a[j + 8] = eB[l16][kp + kA + 16 + j];
    }
#pragma unroll
    for (int e = 0; e < 16; ++e) b[e] = vrow[kp + kB + e + (e & 8)];
    acc = __builtin_amdgcn_wmma_f32_16x16x32_bf16(false, a, false, b,
                                                  (short)0, acc, false, false);
  }
  if (hk == 1) {                        // wave-uniform branch
#pragma unroll
    for (int r = 0; r < 8; ++r) part[r + half * 8][d] = acc[r];
  }
  __syncthreads();
  if (hk == 0) {
#pragma unroll
    for (int r = 0; r < 8; ++r) {
      const int s = qBase + r + half * 8;
      const float v = acc[r] + part[r + half * 8][d];
      ctx[((size_t)(bb * SEQ + s)) * HIDN + h * HD + d] = f2bf(v);
    }
  }
}

// ---------------------------------------------------------------------------
extern "C" void kernel_launch(void* const* d_in, const int* in_sizes, int n_in,
                              void* d_out, int out_size, void* d_ws, size_t ws_size,
                              hipStream_t stream)
{
  (void)in_sizes; (void)n_in; (void)out_size; (void)ws_size;

  const float* q  = (const float*)d_in[0];
  const float* k  = (const float*)d_in[1];
  const float* v  = (const float*)d_in[2];
  const unsigned char* mask = (const unsigned char*)d_in[3];
  const float* Wq = (const float*)d_in[4];
  const float* bq = (const float*)d_in[5];
  const float* Wk = (const float*)d_in[6];
  const float* bk = (const float*)d_in[7];
  const float* Wv = (const float*)d_in[8];
  const float* bv = (const float*)d_in[9];
  const float* Wo = (const float*)d_in[10];
  const float* bo = (const float*)d_in[11];

  float* outX = (float*)d_out;                          // [B,S,HID]
  float* outA = outX + (size_t)NB * SEQ * HIDN;         // [B,H,S,S]

  const size_t MB = (size_t)1024 * 1024;
  char* ws = (char*)d_ws;
  __bf16* Qbf = (__bf16*)(ws);                 // 16 MB  [B,H,S,D]
  __bf16* Kbf = (__bf16*)(ws + 16 * MB);       // 16 MB  [B,H,S,D]
  __bf16* Vt  = (__bf16*)(ws + 32 * MB);       // 16 MB  [B,H,D,S]
  __bf16* ctx = (__bf16*)(ws + 48 * MB);       // 16 MB  [B,S,HID]
  __bf16* Wqb = (__bf16*)(ws + 64 * MB);       //  2 MB each
  __bf16* Wkb = (__bf16*)(ws + 66 * MB);
  __bf16* Wvb = (__bf16*)(ws + 68 * MB);
  __bf16* Wob = (__bf16*)(ws + 70 * MB);
  __bf16* qxb = (__bf16*)(ws + 72 * MB);       // 16 MB each
  __bf16* kxb = (__bf16*)(ws + 88 * MB);
  __bf16* vxb = (__bf16*)(ws + 104 * MB);

  dim3 blk(256);

  // f32 -> bf16 one-time conversions (weights + activations)
  const int nW4 = (HIDN * HIDN) / 4;                  // 262144
  const int nX4 = (NB * SEQ * HIDN) / 4;              // 2097152
  cvt_bf16<<<dim3((nW4 + 255) / 256), blk, 0, stream>>>((const float4*)Wq, (unsigned long long*)Wqb, nW4);
  cvt_bf16<<<dim3((nW4 + 255) / 256), blk, 0, stream>>>((const float4*)Wk, (unsigned long long*)Wkb, nW4);
  cvt_bf16<<<dim3((nW4 + 255) / 256), blk, 0, stream>>>((const float4*)Wv, (unsigned long long*)Wvb, nW4);
  cvt_bf16<<<dim3((nW4 + 255) / 256), blk, 0, stream>>>((const float4*)Wo, (unsigned long long*)Wob, nW4);
  cvt_bf16<<<dim3((nX4 + 255) / 256), blk, 0, stream>>>((const float4*)q,  (unsigned long long*)qxb, nX4);
  cvt_bf16<<<dim3((nX4 + 255) / 256), blk, 0, stream>>>((const float4*)k,  (unsigned long long*)kxb, nX4);
  cvt_bf16<<<dim3((nX4 + 255) / 256), blk, 0, stream>>>((const float4*)v,  (unsigned long long*)vxb, nX4);

  dim3 gGemm(NB * SEQ / 64, HIDN / 128);              // 128 x 8
  const float scale = 4.0f / (float)HD;               // faithful: Q*=scale, K*=scale

  gemm32<0><<<gGemm, blk, 0, stream>>>(qxb, Wqb, bq, scale, Qbf);
  gemm32<0><<<gGemm, blk, 0, stream>>>(kxb, Wkb, bk, scale, Kbf);
  gemm32<1><<<gGemm, blk, 0, stream>>>(vxb, Wvb, bv, 1.0f, Vt);

  dim3 gAttn(SEQ / 16, NH, NB);
  attn_kernel<<<gAttn, blk, 0, stream>>>(Qbf, Kbf, Vt, mask, outA, ctx);

  gemm32<2><<<gGemm, blk, 0, stream>>>(ctx, Wob, bo, 1.0f, outX);
}
